// Block_20426864459937
// MI455X (gfx1250) — matrix operand order repaired and checked by reading, not compiled, not run on hardware
//
// CDNA5 / gfx1250 transformer block (LN+MHA+MoE), bf16 WMMA everywhere.
//
// Precision/data-movement rationale (MI455X):
//  * All contractions use v_wmma_f32_16x16x32_bf16 (f32 accumulate). bf16
//    halves weight bandwidth (134MB vs 268MB for expert weights) and the
//    bf16 WMMA shape carries 8x more K per instruction than f32 WMMA.
//  * Weights are converted once per launch to TRANSPOSED bf16 so every
//    WMMA B-fragment is two contiguous 16B loads per lane.
//  * 192MB L2 holds the full bf16 expert weight set, so the GEMM's
//    redundant B reads across M-tiles hit L2, and HBM sees weights once.
//  * GEMM wave tile is 32x64; the K loop is an explicit ping-pong software
//    pipeline (next chunk's 12 b128 loads issued into the alternate
//    register set before the current chunk's 8 WMMAs) + global_prefetch
//    two chunks ahead, so load latency hides under matrix ops.
//  * Attention is flash-style (no TxT scores in memory): WMMA for Q*K^T
//    and P*V, online softmax in f32, P redistributed C-layout -> A-layout
//    through per-wave LDS tiles; V fragments are loaded before the softmax
//    VALU section so their latency hides under it.
// Workspace requirement: ~266 MB.

#include <hip/hip_runtime.h>
#include <hip/hip_bf16.h>
#include <math.h>

#define BB   2
#define TT   1024
#define DD   1024
#define HH   16
#define DHD  64
#define DFF  4096
#define NEXPT 8
#define NTOK (BB*TT)   // 2048
#define CAP  512
#define NEGV (-1.0e30f)

typedef __attribute__((ext_vector_type(16))) __bf16        v16bf;
typedef __attribute__((ext_vector_type(8)))  float         v8f;
typedef __attribute__((ext_vector_type(8)))  unsigned int  v8u;

__device__ __forceinline__ unsigned short f2bf(float f) {
  unsigned int u = __builtin_bit_cast(unsigned int, f);
  u += 0x7FFFu + ((u >> 16) & 1u);          // round-to-nearest-even
  return (unsigned short)(u >> 16);
}

__device__ __forceinline__ v8f wmma_bf16(v8u a, v8u b, v8f c) {
  return __builtin_amdgcn_wmma_f32_16x16x32_bf16(
      false, __builtin_bit_cast(v16bf, a),
      false, __builtin_bit_cast(v16bf, b),
      (short)0, c, false, false);
}

// A-fragment (16x32 bf16, row-major source). p = rowBase + k0 + hi*16.
// Layout: element e -> k = k0 + hi*16 + e (contiguous, matches B).
__device__ __forceinline__ v8u load_frag_a(const unsigned short* p) {
  const uint4* q = (const uint4*)p;
  uint4 x0 = q[0];     // ushorts 0..7
  uint4 x1 = q[1];     // ushorts 8..15
  return (v8u){x0.x, x0.y, x0.z, x0.w, x1.x, x1.y, x1.z, x1.w};
}

// B-fragment (32x16 bf16) from K-contiguous storage. p = colBase + k0 + hi*16.
// Layout: lane column n = lo, element e -> k = k0 + hi*16 + e (contiguous).
__device__ __forceinline__ v8u load_frag_b(const unsigned short* p) {
  const uint4* q = (const uint4*)p;
  uint4 x0 = q[0];
  uint4 x1 = q[1];
  return (v8u){x0.x, x0.y, x0.z, x0.w, x1.x, x1.y, x1.z, x1.w};
}

// ---------------------------------------------------------------------------
// Weight convert + transpose: src f32 [mat][K][N] -> dst bf16 [mat][N][K]
// ---------------------------------------------------------------------------
__global__ void convT_kernel(const float* __restrict__ src,
                             unsigned short* __restrict__ dst,
                             int K, int N, long long total) {
  long long idx = (long long)blockIdx.x * blockDim.x + threadIdx.x;
  if (idx >= total) return;
  long long kn  = (long long)K * N;
  long long mat = idx / kn;
  long long rem = idx - mat * kn;
  int k = (int)(rem / N);
  int n = (int)(rem - (long long)k * N);
  dst[mat * kn + (long long)n * K + k] = f2bf(src[idx]);
}

// ---------------------------------------------------------------------------
// LayerNorm: one block per row. Writes bf16 (and optionally f32).
// ---------------------------------------------------------------------------
__global__ __launch_bounds__(256) void ln_kernel(
    const float* __restrict__ x, const float* __restrict__ g,
    const float* __restrict__ bta, unsigned short* __restrict__ obf,
    float* __restrict__ of32) {
  __shared__ float red[256];
  int row = blockIdx.x;
  const float* xr = x + (long long)row * DD;
  float s = 0.f, s2 = 0.f;
  for (int i = threadIdx.x; i < DD; i += 256) {
    float v = xr[i]; s += v; s2 += v * v;
  }
  red[threadIdx.x] = s; __syncthreads();
  for (int st = 128; st > 0; st >>= 1) {
    if (threadIdx.x < st) red[threadIdx.x] += red[threadIdx.x + st];
    __syncthreads();
  }
  float mean = red[0] * (1.0f / DD);
  __syncthreads();
  red[threadIdx.x] = s2; __syncthreads();
  for (int st = 128; st > 0; st >>= 1) {
    if (threadIdx.x < st) red[threadIdx.x] += red[threadIdx.x + st];
    __syncthreads();
  }
  float var = red[0] * (1.0f / DD) - mean * mean;
  float rstd = rsqrtf(var + 1e-5f);
  for (int i = threadIdx.x; i < DD; i += 256) {
    float v = (xr[i] - mean) * rstd * g[i] + bta[i];
    obf[(long long)row * DD + i] = f2bf(v);
    if (of32) of32[(long long)row * DD + i] = v;
  }
}

// ---------------------------------------------------------------------------
// Generic bf16 WMMA GEMM: C[M,N] = A[M,K] (bf16 row-major) x Bt[N,K] (bf16).
// Block 256 thr = 8 waves; block tile 128(M) x 128(N); wave tile 32 x 64.
// Explicit ping-pong software pipeline: fragments for K-chunk i+1 are
// loaded into the alternate register set before chunk i's 8 WMMAs issue,
// so the s_wait for each chunk's loads lands a full WMMA chain later.
// Requires K divisible by 64 and K >= 128 (true here: 1024, 4096).
// MODE 0: f32 out.  1: f32 out + residual.  2: bf16 out, +bias, relu.
// MODE 3: f32 out, +bias.  blockIdx.z = expert (per-operand strides).
// ---------------------------------------------------------------------------
template <int MODE>
__global__ __launch_bounds__(256) void gemm_bf16_wmma(
    const unsigned short* __restrict__ A, long long strideA,
    const unsigned short* __restrict__ Bt, long long strideB,
    void* __restrict__ Cv, long long strideC,
    const float* __restrict__ bias, long long strideBias,
    const float* __restrict__ resid, int M, int N, int K) {
  int e = blockIdx.z;
  A  += (long long)e * strideA;
  Bt += (long long)e * strideB;
  int lane = threadIdx.x & 31, wave = threadIdx.x >> 5;
  int lo = lane & 15, hi = lane >> 4;
  int wr = wave >> 1, wc = wave & 1;
  int mBase = blockIdx.y * 128 + wr * 32;
  int nBase = blockIdx.x * 128 + wc * 64;

  v8f acc[2][4] = {};
  const unsigned short* ApRow0 = A + (long long)(mBase + lo) * K + hi * 16;
  const unsigned short* ApRow1 = ApRow0 + 16LL * K;
  const unsigned short* Bp = Bt + (long long)(nBase + lo) * K + hi * 16;

  auto loadA2 = [&](v8u* a, int k) {
    a[0] = load_frag_a(ApRow0 + k);
    a[1] = load_frag_a(ApRow1 + k);
  };
  auto loadB4 = [&](v8u* bb, int k) {
#pragma unroll
    for (int nt = 0; nt < 4; ++nt)
      bb[nt] = load_frag_b(Bp + (long long)(nt * 16) * K + k);
  };
  auto mmac = [&](v8u* a, v8u* bb) {
#pragma unroll
    for (int mt = 0; mt < 2; ++mt)
#pragma unroll
      for (int nt = 0; nt < 4; ++nt)
        acc[mt][nt] = wmma_bf16(a[mt], bb[nt], acc[mt][nt]);
  };

  v8u a0[2], b0[4], a1[2], b1[4];
  loadA2(a0, 0);
  loadB4(b0, 0);
  int k0 = 0;
  for (; k0 + 64 < K; k0 += 64) {
    loadA2(a1, k0 + 32);
    loadB4(b1, k0 + 32);
    if (k0 + 128 < K) {                    // pull next-next chunk toward L0/L2
      __builtin_prefetch(ApRow0 + k0 + 128, 0, 0);
      __builtin_prefetch(ApRow1 + k0 + 128, 0, 0);
#pragma unroll
      for (int nt = 0; nt < 4; ++nt)
        __builtin_prefetch(Bp + (long long)(nt * 16) * K + k0 + 128, 0, 0);
    }
    mmac(a0, b0);                          // consumes chunk k0 (loaded earlier)
    loadA2(a0, k0 + 64);
    loadB4(b0, k0 + 64);
    mmac(a1, b1);                          // consumes chunk k0+32
  }
  // tail: k0 == K-64; a0/b0 hold chunk k0
  loadA2(a1, k0 + 32);
  loadB4(b1, k0 + 32);
  mmac(a0, b0);
  mmac(a1, b1);

#pragma unroll
  for (int mt = 0; mt < 2; ++mt) {
#pragma unroll
    for (int nt = 0; nt < 4; ++nt) {
      int n = nBase + nt * 16 + lo;
      float bv = (MODE == 2 || MODE == 3) ? bias[(long long)e * strideBias + n] : 0.f;
#pragma unroll
      for (int r = 0; r < 8; ++r) {
        int m = mBase + mt * 16 + hi * 8 + r;
        float v = acc[mt][nt][r] + bv;
        long long ci = (long long)e * strideC + (long long)m * N + n;
        if (MODE == 0)      ((float*)Cv)[ci] = v;
        else if (MODE == 1) ((float*)Cv)[ci] = v + resid[(long long)m * N + n];
        else if (MODE == 2) ((unsigned short*)Cv)[ci] = f2bf(fmaxf(v, 0.f));
        else                ((float*)Cv)[ci] = v;
      }
    }
  }
}

// ---------------------------------------------------------------------------
// RoPE + layout shuffle. qkv f32 [tok][3*D] -> q,k bf16 [bh][t][dh],
// v bf16 TRANSPOSED [bh][dh][t] (so P*V B-fragments are contiguous).
// ---------------------------------------------------------------------------
__global__ void rope_kernel(const float* __restrict__ qkv,
                            unsigned short* __restrict__ qb,
                            unsigned short* __restrict__ kb,
                            unsigned short* __restrict__ vtb) {
  int idx = blockIdx.x * blockDim.x + threadIdx.x;  // NTOK*HH*32 threads
  int i = idx & 31;
  int tmp = idx >> 5;
  int h = tmp & (HH - 1);
  int tk = tmp >> 4;                 // token id
  int t = tk & (TT - 1);
  int b = tk >> 10;
  const float* base = qkv + (long long)tk * 3 * DD;
  float q1 = base[0 * DD + h * DHD + i],      q2 = base[0 * DD + h * DHD + 32 + i];
  float k1 = base[1 * DD + h * DHD + i],      k2 = base[1 * DD + h * DHD + 32 + i];
  float v1 = base[2 * DD + h * DHD + i],      v2 = base[2 * DD + h * DHD + 32 + i];
  float inv = __expf(-(float)(2 * i) * (9.210340371976184f / 64.f));
  float ang = (float)t * inv;
  float sn = __sinf(ang), cs = __cosf(ang);
  int bh = b * HH + h;
  long long qo = ((long long)bh * TT + t) * DHD;
  qb[qo + i]      = f2bf(q1 * cs - q2 * sn);
  qb[qo + 32 + i] = f2bf(q2 * cs + q1 * sn);
  kb[qo + i]      = f2bf(k1 * cs - k2 * sn);
  kb[qo + 32 + i] = f2bf(k2 * cs + k1 * sn);
  vtb[((long long)bh * DHD + i) * TT + t]      = f2bf(v1);
  vtb[((long long)bh * DHD + 32 + i) * TT + t] = f2bf(v2);
}

// ---------------------------------------------------------------------------
// Flash attention. Block = 4 waves, 64 query rows per block, one (b,h).
// Wave owns 16 rows: S=Q*K^T via WMMA, online softmax, O += P*V via WMMA.
// P goes C-layout -> A-layout through a per-wave LDS tile. V fragments are
// loaded before the softmax VALU section so their latency hides under it.
// ---------------------------------------------------------------------------
__global__ __launch_bounds__(128) void attn_flash(
    const unsigned short* __restrict__ qb, const unsigned short* __restrict__ kb,
    const unsigned short* __restrict__ vtb, unsigned short* __restrict__ ao) {
  __shared__ __align__(16) unsigned short Pl[4 * 16 * 64];
  int bh = blockIdx.x;
  int h = bh & (HH - 1);
  int b = bh >> 4;
  int lane = threadIdx.x & 31, wave = threadIdx.x >> 5;
  int lo = lane & 15, hi = lane >> 4;
  int qm0 = blockIdx.y * 64;
  int qrowA = qm0 + wave * 16 + lo;  // A-layout row for this lane

  const unsigned short* Qp = qb + ((long long)bh * TT + qrowA) * DHD;
  v8u qa[2];
#pragma unroll
  for (int c = 0; c < 2; ++c) qa[c] = load_frag_a(Qp + c * 32 + hi * 16);

  float mrow[8], lrow[8];
#pragma unroll
  for (int r = 0; r < 8; ++r) { mrow[r] = -3.0e38f; lrow[r] = 0.f; }
  v8f o[4] = {};
  unsigned short* myP = Pl + wave * 1024;
  int nkt = qm0 / 64 + 1;

  for (int kt = 0; kt < nkt; ++kt) {
    int s0 = kt * 64;
    v8f s[4] = {};
    // ---- batch K loads; issue V loads too (consumed after softmax) ----
    v8u kf[4][2], vf[4][2];
#pragma unroll
    for (int nt = 0; nt < 4; ++nt) {
      const unsigned short* Kp =
          kb + ((long long)bh * TT + s0 + nt * 16 + lo) * DHD + hi * 16;
      kf[nt][0] = load_frag_b(Kp);
      kf[nt][1] = load_frag_b(Kp + 32);
    }
#pragma unroll
    for (int nt = 0; nt < 4; ++nt) {
      const unsigned short* Vp =
          vtb + ((long long)bh * DHD + nt * 16 + lo) * TT + s0 + hi * 16;
      vf[nt][0] = load_frag_b(Vp);
      vf[nt][1] = load_frag_b(Vp + 32);
    }
    // ---- S = Q*K^T ----
#pragma unroll
    for (int nt = 0; nt < 4; ++nt)
#pragma unroll
      for (int c = 0; c < 2; ++c)
        s[nt] = wmma_bf16(qa[c], kf[nt][c], s[nt]);

    float pmat[4][8];
#pragma unroll
    for (int r = 0; r < 8; ++r) {
      int trow = qm0 + wave * 16 + hi * 8 + r;
      float mx = -3.0e38f;
#pragma unroll
      for (int nt = 0; nt < 4; ++nt) {
        int sc = s0 + nt * 16 + lo;
        float v = s[nt][r] * 0.125f;          // 1/sqrt(64)
        v = (sc > trow) ? NEGV : v;           // causal mask
        pmat[nt][r] = v;
        mx = fmaxf(mx, v);
      }
#pragma unroll
      for (int d = 1; d < 16; d <<= 1) mx = fmaxf(mx, __shfl_xor(mx, d, 32));
      float mnew = fmaxf(mrow[r], mx);
      float alpha = __expf(mrow[r] - mnew);
      float rsum = 0.f;
#pragma unroll
      for (int nt = 0; nt < 4; ++nt) {
        float p = __expf(pmat[nt][r] - mnew);
        pmat[nt][r] = p;
        rsum += p;
      }
#pragma unroll
      for (int d = 1; d < 16; d <<= 1) rsum += __shfl_xor(rsum, d, 32);
      lrow[r] = lrow[r] * alpha + rsum;
      mrow[r] = mnew;
#pragma unroll
      for (int nt = 0; nt < 4; ++nt) o[nt][r] *= alpha;
    }
    // C-layout P -> LDS (row-major 16x64 per wave)
#pragma unroll
    for (int r = 0; r < 8; ++r)
#pragma unroll
      for (int nt = 0; nt < 4; ++nt)
        myP[(hi * 8 + r) * 64 + nt * 16 + lo] = f2bf(pmat[nt][r]);
    __syncthreads();
    // ---- O += P @ V (V fragments already resident) ----
    v8u pa[2];
#pragma unroll
    for (int c = 0; c < 2; ++c)
      pa[c] = load_frag_a(myP + lo * 64 + c * 32 + hi * 16);
#pragma unroll
    for (int c = 0; c < 2; ++c)
#pragma unroll
      for (int nt = 0; nt < 4; ++nt)
        o[nt] = wmma_bf16(pa[c], vf[nt][c], o[nt]);
    __syncthreads();
  }
  // epilogue: divide by l, write ao[b][t][h*64+dh] bf16
#pragma unroll
  for (int r = 0; r < 8; ++r) {
    float inv = 1.0f / lrow[r];
    int t = qm0 + wave * 16 + hi * 8 + r;
#pragma unroll
    for (int nt = 0; nt < 4; ++nt) {
      int dcol = h * DHD + nt * 16 + lo;
      ao[((long long)(b * TT + t)) * DD + dcol] = f2bf(o[nt][r] * inv);
    }
  }
}

// ---------------------------------------------------------------------------
// Router: noisy top-2 per token (thread per token; w_rl/w_rn broadcast via L2).
// ---------------------------------------------------------------------------
__global__ void router_kernel(const float* __restrict__ h2,
                              const float* __restrict__ wrl, const float* __restrict__ brl,
                              const float* __restrict__ wrn, const float* __restrict__ brn,
                              const float* __restrict__ noise,
                              int* __restrict__ ix, float* __restrict__ pv,
                              int* __restrict__ tokE) {
  int t = blockIdx.x * blockDim.x + threadIdx.x;
  if (t >= NTOK) return;
  float al[NEXPT], an[NEXPT];
#pragma unroll
  for (int e = 0; e < NEXPT; ++e) { al[e] = brl[e]; an[e] = brn[e]; }
  const float* hr = h2 + (long long)t * DD;
  for (int d = 0; d < DD; ++d) {
    float hv = hr[d];
#pragma unroll
    for (int e = 0; e < NEXPT; ++e) {
      al[e] += hv * wrl[d * NEXPT + e];
      an[e] += hv * wrn[d * NEXPT + e];
    }
  }
  float noisy[NEXPT];
#pragma unroll
  for (int e = 0; e < NEXPT; ++e) {
    float sp = an[e];
    sp = (sp > 20.f) ? sp : logf(1.f + __expf(sp));   // softplus
    noisy[e] = al[e] + noise[t * NEXPT + e] * sp;
  }
  int i0 = 0; float v0 = noisy[0];
#pragma unroll
  for (int e = 1; e < NEXPT; ++e) if (noisy[e] > v0) { v0 = noisy[e]; i0 = e; }
  int i1 = -1; float v1 = -3.0e38f;
#pragma unroll
  for (int e = 0; e < NEXPT; ++e)
    if (e != i0 && noisy[e] > v1) { v1 = noisy[e]; i1 = e; }
  float e1 = __expf(v1 - v0);
  float den = 1.f + e1;
  ix[t * 2] = i0;  ix[t * 2 + 1] = i1;
  pv[t * 2] = 1.f / den;  pv[t * 2 + 1] = e1 / den;
  tokE[t * 2] = -1;  tokE[t * 2 + 1] = -1;
}

// Capacity assignment: thread per expert, stable in-order scan (matches
// jnp stable argsort of !mask: accepted = first <=CAP assigned tokens).
__global__ void assign_kernel(const int* __restrict__ ix, const float* __restrict__ pv,
                              int* __restrict__ sel, int* __restrict__ tokE,
                              int* __restrict__ tokS, float* __restrict__ tokG) {
  int e = threadIdx.x;
  if (e >= NEXPT) return;
  int cnt = 0;
  for (int t = 0; t < NTOK; ++t) {
    int j = (ix[t * 2] == e) ? 0 : ((ix[t * 2 + 1] == e) ? 1 : -1);
    if (j >= 0 && cnt < CAP) {
      sel[e * CAP + cnt] = t;
      tokE[t * 2 + j] = e;
      tokS[t * 2 + j] = cnt;
      tokG[t * 2 + j] = pv[t * 2 + j];
      cnt++;
    }
  }
  for (; cnt < CAP; ++cnt) sel[e * CAP + cnt] = 0;  // harmless filler (gate 0)
}

// Gather selected tokens into per-expert dense A matrices (u32 = 2 bf16).
__global__ void gather_kernel(const unsigned short* __restrict__ h2b,
                              const int* __restrict__ sel,
                              unsigned short* __restrict__ xe) {
  long long idx = (long long)blockIdx.x * blockDim.x + threadIdx.x;
  const long long total = (long long)NEXPT * CAP * (DD / 2);
  if (idx >= total) return;
  int dp = (int)(idx % (DD / 2));
  long long r = idx / (DD / 2);
  int tok = sel[r];
  ((unsigned int*)xe)[r * (DD / 2) + dp] =
      ((const unsigned int*)h2b)[(long long)tok * (DD / 2) + dp];
}

// out = x2 + sum_j gate_j * o[e_j][slot_j][:]   (no atomics: inverse map)
__global__ void final_kernel(const float* __restrict__ x2,
                             const int* __restrict__ tokE, const int* __restrict__ tokS,
                             const float* __restrict__ tokG,
                             const float* __restrict__ of, float* __restrict__ out) {
  long long idx = (long long)blockIdx.x * blockDim.x + threadIdx.x;
  if (idx >= (long long)NTOK * DD) return;
  int d = (int)(idx & (DD - 1));
  long long t = idx >> 10;
  float acc = x2[idx];
#pragma unroll
  for (int j = 0; j < 2; ++j) {
    int e = tokE[t * 2 + j];
    if (e >= 0) {
      int s = tokS[t * 2 + j];
      acc += tokG[t * 2 + j] * of[(((long long)e * CAP + s) * DD) + d];
    }
  }
  out[idx] = acc;
}

// ---------------------------------------------------------------------------
extern "C" void kernel_launch(void* const* d_in, const int* in_sizes, int n_in,
                              void* d_out, int out_size, void* d_ws, size_t ws_size,
                              hipStream_t stream) {
  const float* x      = (const float*)d_in[0];
  const float* noise  = (const float*)d_in[1];
  const float* ln1_g  = (const float*)d_in[2];
  const float* ln1_b  = (const float*)d_in[3];
  const float* ln2_g  = (const float*)d_in[4];
  const float* ln2_b  = (const float*)d_in[5];
  const float* w_qkv  = (const float*)d_in[6];
  const float* w_proj = (const float*)d_in[7];
  const float* w_rl   = (const float*)d_in[8];
  const float* b_rl   = (const float*)d_in[9];
  const float* w_rn   = (const float*)d_in[10];
  const float* b_rn   = (const float*)d_in[11];
  const float* w1     = (const float*)d_in[12];
  const float* b1     = (const float*)d_in[13];
  const float* w2     = (const float*)d_in[14];
  const float* b2     = (const float*)d_in[15];
  float* out = (float*)d_out;

  char* ws = (char*)d_ws;
  size_t off = 0;
  auto alloc = [&](size_t bytes) -> void* {
    off = (off + 255) & ~(size_t)255;
    void* p = ws + off;
    off += bytes;
    return p;
  };
  unsigned short* wqkvT = (unsigned short*)alloc((size_t)3072 * 1024 * 2);
  unsigned short* wprojT= (unsigned short*)alloc((size_t)1024 * 1024 * 2);
  unsigned short* w1T   = (unsigned short*)alloc((size_t)NEXPT * DD * DFF * 2);
  unsigned short* w2T   = (unsigned short*)alloc((size_t)NEXPT * DFF * DD * 2);
  unsigned short* h1b   = (unsigned short*)alloc((size_t)NTOK * DD * 2);
  float*          qkvf  = (float*)alloc((size_t)NTOK * 3 * DD * 4);
  unsigned short* qb    = (unsigned short*)alloc((size_t)BB * HH * TT * DHD * 2);
  unsigned short* kb    = (unsigned short*)alloc((size_t)BB * HH * TT * DHD * 2);
  unsigned short* vtb   = (unsigned short*)alloc((size_t)BB * HH * TT * DHD * 2);
  unsigned short* aob   = (unsigned short*)alloc((size_t)NTOK * DD * 2);
  float*          x2f   = (float*)alloc((size_t)NTOK * DD * 4);
  float*          h2f   = (float*)alloc((size_t)NTOK * DD * 4);
  unsigned short* h2b   = (unsigned short*)alloc((size_t)NTOK * DD * 2);
  int*            ixb   = (int*)alloc((size_t)NTOK * 2 * 4);
  float*          pvb   = (float*)alloc((size_t)NTOK * 2 * 4);
  int*            tokE  = (int*)alloc((size_t)NTOK * 2 * 4);
  int*            tokS  = (int*)alloc((size_t)NTOK * 2 * 4);
  float*          tokG  = (float*)alloc((size_t)NTOK * 2 * 4);
  int*            sel   = (int*)alloc((size_t)NEXPT * CAP * 4);
  unsigned short* xeb   = (unsigned short*)alloc((size_t)NEXPT * CAP * DD * 2);
  unsigned short* t1b   = (unsigned short*)alloc((size_t)NEXPT * CAP * DFF * 2);
  float*          ofp   = (float*)alloc((size_t)NEXPT * CAP * DD * 4);
  (void)ws_size; (void)in_sizes; (void)n_in; (void)out_size;

  // 1) weights -> transposed bf16
  {
    long long tot = (long long)1024 * 3072;
    convT_kernel<<<(unsigned)((tot + 255) / 256), 256, 0, stream>>>(w_qkv, wqkvT, 1024, 3072, tot);
    tot = (long long)1024 * 1024;
    convT_kernel<<<(unsigned)((tot + 255) / 256), 256, 0, stream>>>(w_proj, wprojT, 1024, 1024, tot);
    tot = (long long)NEXPT * DD * DFF;
    convT_kernel<<<(unsigned)((tot + 255) / 256), 256, 0, stream>>>(w1, w1T, DD, DFF, tot);
    convT_kernel<<<(unsigned)((tot + 255) / 256), 256, 0, stream>>>(w2, w2T, DFF, DD, tot);
  }
  // 2) LN1
  ln_kernel<<<NTOK, 256, 0, stream>>>(x, ln1_g, ln1_b, h1b, nullptr);
  // 3) qkv = h1 @ w_qkv   (2048 x 3072 x 1024)
  gemm_bf16_wmma<0><<<dim3(3072 / 128, NTOK / 128, 1), 256, 0, stream>>>(
      h1b, 0, wqkvT, 0, qkvf, 0, nullptr, 0, nullptr, NTOK, 3072, 1024);
  // 4) RoPE + split/transpose
  rope_kernel<<<(NTOK * HH * 32) / 256, 256, 0, stream>>>(qkvf, qb, kb, vtb);
  // 5) flash attention -> ao (bf16)
  attn_flash<<<dim3(BB * HH, TT / 64), 128, 0, stream>>>(qb, kb, vtb, aob);
  // 6) x2 = x + ao @ w_proj
  gemm_bf16_wmma<1><<<dim3(1024 / 128, NTOK / 128, 1), 256, 0, stream>>>(
      aob, 0, wprojT, 0, x2f, 0, nullptr, 0, x, NTOK, 1024, 1024);
  // 7) LN2 (f32 for router, bf16 for experts)
  ln_kernel<<<NTOK, 256, 0, stream>>>(x2f, ln2_g, ln2_b, h2b, h2f);
  // 8) router
  router_kernel<<<(NTOK + 255) / 256, 256, 0, stream>>>(
      h2f, w_rl, b_rl, w_rn, b_rn, noise, ixb, pvb, tokE);
  // 9) capacity assignment
  assign_kernel<<<1, 32, 0, stream>>>(ixb, pvb, sel, tokE, tokS, tokG);
  // 10) gather expert inputs
  {
    long long tot = (long long)NEXPT * CAP * (DD / 2);
    gather_kernel<<<(unsigned)((tot + 255) / 256), 256, 0, stream>>>(h2b, sel, xeb);
  }
  // 11) t1 = relu(xe @ w1 + b1)  per expert (512 x 4096 x 1024), bf16 out
  gemm_bf16_wmma<2><<<dim3(DFF / 128, CAP / 128, NEXPT), 256, 0, stream>>>(
      xeb, (long long)CAP * DD, w1T, (long long)DFF * DD, t1b,
      (long long)CAP * DFF, b1, DFF, nullptr, CAP, DFF, DD);
  // 12) o = t1 @ w2 + b2  per expert (512 x 1024 x 4096), f32 out
  gemm_bf16_wmma<3><<<dim3(DD / 128, CAP / 128, NEXPT), 256, 0, stream>>>(
      t1b, (long long)CAP * DFF, w2T, (long long)DD * DFF, ofp,
      (long long)CAP * DD, b2, DD, nullptr, CAP, DD, DFF);
  // 13) out = x2 + gated expert updates
  {
    long long tot = (long long)NTOK * DD;
    final_kernel<<<(unsigned)((tot + 255) / 256), 256, 0, stream>>>(
        x2f, tokE, tokS, tokG, ofp, out);
  }
}
